// MemoryAttention_7730941133115
// MI455X (gfx1250) — matrix operand order repaired and checked by reading, not compiled, not run on hardware
//
#include <hip/hip_runtime.h>
#include <math.h>

// MI455X / gfx1250: wave32, WMMA 16x16x32 bf16 (f32 accumulate), TDM mask DMA,
// DPP16 lane reductions (no ds_bpermute).
typedef __attribute__((ext_vector_type(16))) __bf16 v16bf;
typedef __attribute__((ext_vector_type(8)))  __bf16 v8bf;
typedef __attribute__((ext_vector_type(8)))  float  v8f;
typedef __attribute__((ext_vector_type(4)))  float  v4f;
typedef __attribute__((ext_vector_type(4)))  unsigned int u32x4;
typedef __attribute__((ext_vector_type(8)))  unsigned int u32x8;

#define B_    16
#define S_    2048
#define H_    1024
#define N_    2048          // n_examples * mem_len = 16 * 128
#define QSTR  1032          // bf16 elements per Q row in LDS (padded, 16B-multiple)
#define PSTR  136           // bf16 elements per P row in LDS (padded, 16B-multiple)

// ---- 16-lane (half-wave) reductions, pure VALU via DPP16 ----
// xor1 = quad_perm[1,0,3,2] (0xB1), xor2 = quad_perm[2,3,0,1] (0x4E),
// xor4 ~ row_half_mirror (0x141), xor8 ~ row_mirror (0x140).
__device__ __forceinline__ float dpp_max16(float x) {
    int t;
    t = __builtin_amdgcn_update_dpp(0, __float_as_int(x), 0xB1,  0xF, 0xF, true);
    x = fmaxf(x, __int_as_float(t));
    t = __builtin_amdgcn_update_dpp(0, __float_as_int(x), 0x4E,  0xF, 0xF, true);
    x = fmaxf(x, __int_as_float(t));
    t = __builtin_amdgcn_update_dpp(0, __float_as_int(x), 0x141, 0xF, 0xF, true);
    x = fmaxf(x, __int_as_float(t));
    t = __builtin_amdgcn_update_dpp(0, __float_as_int(x), 0x140, 0xF, 0xF, true);
    x = fmaxf(x, __int_as_float(t));
    return x;
}
__device__ __forceinline__ float dpp_sum16(float x) {
    int t;
    t = __builtin_amdgcn_update_dpp(0, __float_as_int(x), 0xB1,  0xF, 0xF, true);
    x += __int_as_float(t);
    t = __builtin_amdgcn_update_dpp(0, __float_as_int(x), 0x4E,  0xF, 0xF, true);
    x += __int_as_float(t);
    t = __builtin_amdgcn_update_dpp(0, __float_as_int(x), 0x141, 0xF, 0xF, true);
    x += __int_as_float(t);
    t = __builtin_amdgcn_update_dpp(0, __float_as_int(x), 0x140, 0xF, 0xF, true);
    x += __int_as_float(t);
    return x;
}

__global__ __launch_bounds__(256)
void memory_attention_gfx1250(const float* __restrict__ seq,
                              const float* __restrict__ mem,
                              const int*   __restrict__ mmask,
                              float* __restrict__ out)
{
    const int b    = blockIdx.y;
    const int m0   = blockIdx.x * 16;         // Q-tile row base
    const int tid  = threadIdx.x;
    const int wave = tid >> 5;                // 8 waves
    const int lane = tid & 31;
    const int half = lane >> 4;               // wave32 half (0/1)
    const int lcol = lane & 15;

    const float* seqB = seq + (size_t)b * S_ * H_;
    const float* memB = mem + (size_t)b * N_ * H_;
    const int*   mkB  = mmask + b * N_;

    __shared__ alignas(16) __bf16 Qs[16 * QSTR];   // Q tile, bf16          (33.0 KB)
    __shared__ alignas(16) __bf16 Ps[16 * PSTR];   // probability chunk     ( 4.3 KB)
    __shared__ alignas(16) float  fmv[N_];         // mask mult, n-indexed  ( 8.0 KB)
    __shared__ alignas(16) float  fm2[2 * 1024];   // mask mult, [c][k]     ( 8.0 KB)
    __shared__ alignas(16) int    mstage[N_];      // TDM-landed raw mask   ( 8.0 KB)
    __shared__ float red_m[8][16];                 // per-wave row max
    __shared__ float red_s[8][16];                 // per-wave row sum
    __shared__ float Mrow[16];                     // running row max
    __shared__ float Lrow[16];                     // running row denom
    __shared__ float Srow[16];                     // per-chunk rescale

    // ---- TDM: DMA the 8KB mask row into LDS (wave 0 issues; EXEC ignored) ----
    if (wave == 0) {
        const unsigned long long ga = (unsigned long long)(uintptr_t)mkB;
        const unsigned lds_addr = (unsigned)(uintptr_t)(&mstage[0]); // flat LDS: low 32b = offset
        u32x4 g0;
        g0[0] = 1u;                                       // count=1, user D#
        g0[1] = lds_addr;                                 // lds_addr
        g0[2] = (unsigned)(ga & 0xffffffffu);             // global_addr[31:0]
        g0[3] = (unsigned)((ga >> 32) & 0x01ffffffu)      // global_addr[56:32]
              | (2u << 30);                               // type = 2 ("image")
        u32x8 g1;
        g1[0] = (2u << 16);                               // data_size = 4B
        g1[1] = ((unsigned)N_ & 0xffffu) << 16;           // tensor_dim0[15:0]
        g1[2] = ((unsigned)N_ >> 16) | (1u << 16);        // tensor_dim0[31:16] | tensor_dim1[15:0]=1
        g1[3] = ((unsigned)N_ & 0xffffu) << 16;           // tile_dim0 = 2048
        g1[4] = 0u;                                       // tile_dim1/2 unused (1-D tile)
        g1[5] = (unsigned)N_;                             // tensor_dim0_stride
        g1[6] = 0u;
        g1[7] = 0u;
        asm volatile("tensor_load_to_lds %0, %1"
                     :: "s"(g0), "s"(g1) : "memory");
        __builtin_amdgcn_s_wait_tensorcnt(0);             // s_wait_tensorcnt 0
    }

    // ---- stage Q tile (f32 -> bf16), init row stats (overlaps with TDM) ----
    for (int i = tid; i < 16 * H_; i += 256) {
        const int m = i >> 10, h = i & (H_ - 1);
        Qs[m * QSTR + h] = (__bf16)seqB[(size_t)(m0 + m) * H_ + h];
    }
    if (tid < 16) { Mrow[tid] = -__builtin_inff(); Lrow[tid] = 0.0f; }
    __syncthreads();

    // ---- mask int -> multiplier tables (fmv: n-indexed, fm2: [n&1][n>>1]) ----
    for (int i = tid; i < N_; i += 256) {
        const float f = (mstage[i] != 0) ? 1.0f : 0.0f;
        fmv[i] = f;
        fm2[(i & 1) * 1024 + (i >> 1)] = f;
    }
    __syncthreads();

    v8f acc[8] = {};                               // 16 x 128 f32 output slice

    const __bf16* qrow = Qs + lcol * QSTR;         // A-frag row (M = lcol)
    const __bf16* prow = Ps + lcol * PSTR;

    #pragma unroll 1
    for (int chunk = 0; chunk < N_ / 128; ++chunk) {
        // ---------- score tile: sc = Q(16x1024) x Kview(1024 x 16 cols) ----------
        const int ncol = chunk * 128 + wave * 16 + lcol;  // this lane's score column
        const float* fmrow = fm2 + (ncol >> 10) * 1024;   // mask table for this column
        v8f sc = {};

        if (chunk + 1 < N_ / 128)                          // gfx1250 global_prefetch_b8
            __builtin_prefetch(memB + (size_t)tid * N_ + (chunk + 1) * 128, 0, 1);

        #pragma unroll 2
        for (int ks = 0; ks < H_ / 32; ++ks) {             // 32 K-steps of 32
            const int k0 = ks * 32;
            // A fragment (16x32 bf16 layout): two aligned 16B LDS loads
            v8bf alo = *(const v8bf*)(qrow + k0 + half * 16);
            v8bf ahi = *(const v8bf*)(qrow + k0 + half * 16 + 8);
            v16bf a = __builtin_shufflevector(alo, ahi,
                0,1,2,3,4,5,6,7,8,9,10,11,12,13,14,15);
            // B fragment (32x16 bf16): K = 16*half + i, mask-by-zero inline
            const int kb = k0 + half * 16;
            const v4f mva = *(const v4f*)(fmrow + kb);
            const v4f mvb = *(const v4f*)(fmrow + kb + 4);
            const v4f mvc = *(const v4f*)(fmrow + kb + 8);
            const v4f mvd = *(const v4f*)(fmrow + kb + 12);
            v16bf bb;
            #pragma unroll
            for (int i = 0; i < 4; ++i) {
                bb[i]      = (__bf16)(memB[(size_t)(kb + i)      * N_ + ncol] * mva[i]);
                bb[4 + i]  = (__bf16)(memB[(size_t)(kb + 4 + i)  * N_ + ncol] * mvb[i]);
                bb[8 + i]  = (__bf16)(memB[(size_t)(kb + 8 + i)  * N_ + ncol] * mvc[i]);
                bb[12 + i] = (__bf16)(memB[(size_t)(kb + 12 + i) * N_ + ncol] * mvd[i]);
            }
            sc = __builtin_amdgcn_wmma_f32_16x16x32_bf16(
                     false, a, false, bb, (short)0, sc, false, false);
        }

        // ---------- per-wave row max (DPP16, pure VALU) ----------
        #pragma unroll
        for (int r = 0; r < 8; ++r) {
            const float mx = dpp_max16(sc[r]);
            if (lcol == 0) red_m[wave][r + 8 * half] = mx;
        }
        __syncthreads();

        // ---------- merge row stats across waves (online softmax) ----------
        if (tid < 16) {
            float cm = red_m[0][tid];
            #pragma unroll
            for (int w2 = 1; w2 < 8; ++w2) cm = fmaxf(cm, red_m[w2][tid]);
            const float mold = Mrow[tid];
            const float mnew = fmaxf(mold, cm);
            const float s    = __expf(mold - mnew);   // exp(-inf) = 0 first time
            Mrow[tid] = mnew;
            Srow[tid] = s;
            Lrow[tid] *= s;
        }
        __syncthreads();

        // ---------- probabilities -> LDS (bf16), per-wave row sums ----------
        #pragma unroll
        for (int r = 0; r < 8; ++r) {
            const int row = r + 8 * half;
            const float p = __expf(sc[r] - Mrow[row]);
            Ps[row * PSTR + wave * 16 + lcol] = (__bf16)p;
            const float su = dpp_sum16(p);
            if (lcol == 0) red_s[wave][row] = su;
        }
        __syncthreads();

        // ---------- L update, rescale acc, acc += P(16x128) x V(128 x Hslice) ----------
        if (tid < 16) {
            float cs = red_s[0][tid];
            #pragma unroll
            for (int w2 = 1; w2 < 8; ++w2) cs += red_s[w2][tid];
            Lrow[tid] += cs;
        }
        #pragma unroll
        for (int r = 0; r < 8; ++r) {
            const float s = Srow[r + 8 * half];
            #pragma unroll
            for (int t = 0; t < 8; ++t) acc[t][r] *= s;
        }
        #pragma unroll 1
        for (int ks = 0; ks < 4; ++ks) {
            const int k0 = ks * 32;
            v8bf plo = *(const v8bf*)(prow + k0 + half * 16);
            v8bf phi = *(const v8bf*)(prow + k0 + half * 16 + 8);
            v16bf pa = __builtin_shufflevector(plo, phi,
                0,1,2,3,4,5,6,7,8,9,10,11,12,13,14,15);
            const int nbase = chunk * 128 + k0 + half * 16;   // V rows = K dim
            const v4f ma = *(const v4f*)(fmv + nbase);
            const v4f mb = *(const v4f*)(fmv + nbase + 4);
            const v4f mc = *(const v4f*)(fmv + nbase + 8);
            const v4f md = *(const v4f*)(fmv + nbase + 12);
            #pragma unroll 2
            for (int t = 0; t < 8; ++t) {
                const int hcol = wave * 128 + t * 16 + lcol;
                v16bf vb;
                #pragma unroll
                for (int i = 0; i < 4; ++i) {
                    vb[i]      = (__bf16)(memB[(size_t)(nbase + i)      * H_ + hcol] * ma[i]);
                    vb[4 + i]  = (__bf16)(memB[(size_t)(nbase + 4 + i)  * H_ + hcol] * mb[i]);
                    vb[8 + i]  = (__bf16)(memB[(size_t)(nbase + 8 + i)  * H_ + hcol] * mc[i]);
                    vb[12 + i] = (__bf16)(memB[(size_t)(nbase + 12 + i) * H_ + hcol] * md[i]);
                }
                acc[t] = __builtin_amdgcn_wmma_f32_16x16x32_bf16(
                             false, pa, false, vb, (short)0, acc[t], false, false);
            }
        }
    }

    __syncthreads();

    // ---------- epilogue: out = seq + acc / L ----------
    #pragma unroll
    for (int t = 0; t < 8; ++t) {
        const int hcol = wave * 128 + t * 16 + lcol;
        #pragma unroll
        for (int r = 0; r < 8; ++r) {
            const int row = r + 8 * half;
            const float l = Lrow[row];
            const size_t idx = (size_t)(m0 + row) * H_ + hcol;
            out[(size_t)b * S_ * H_ + idx] = seqB[idx] + acc[t][r] / l;
        }
    }
}

extern "C" void kernel_launch(void* const* d_in, const int* in_sizes, int n_in,
                              void* d_out, int out_size, void* d_ws, size_t ws_size,
                              hipStream_t stream) {
    const float* seq  = (const float*)d_in[0];
    // d_in[1] = attention_mask: unused by the reference computation
    const float* mem  = (const float*)d_in[2];
    const int*   mmsk = (const int*)d_in[3];
    float* out = (float*)d_out;

    dim3 grid(S_ / 16, B_);
    memory_attention_gfx1250<<<grid, 256, 0, stream>>>(seq, mem, mmsk, out);

    (void)in_sizes; (void)n_in; (void)out_size; (void)d_ws; (void)ws_size;
}
